// Att_channel_38259568673405
// MI455X (gfx1250) — matrix operand order repaired and checked by reading, not compiled, not run on hardware
//
#include <hip/hip_runtime.h>

typedef unsigned short u16;
typedef __attribute__((ext_vector_type(16))) __bf16 bf16x16;
typedef __attribute__((ext_vector_type(8)))  float  f32x8;
typedef __attribute__((ext_vector_type(4)))  unsigned int u32x4;

struct Pack32 { u32x4 lo, hi; };

#define B_  8
#define L_  896
#define E_  820
#define EP_ 832
#define H_  41
#define DH_ 20
#define IM_ 2304
#define M_  7168            // B_*L_
#define LC_ 128             // L_/CB
#define CB_ 7
#define SB_ 2
#define WS_ 5
#define TOPK_ 16
#define SCALE_ 0.22360679774997896f   // 1/sqrt(20)

#if defined(__has_builtin)
# if __has_builtin(__builtin_amdgcn_global_load_async_to_lds_b128)
#  define HAVE_ASYNC_LDS 1
# endif
#endif

typedef int v4i_ __attribute__((vector_size(16)));
typedef __attribute__((address_space(1))) v4i_ gv4i;
typedef __attribute__((address_space(3))) v4i_ lv4i;

__device__ __forceinline__ void async_copy16(void* lds_dst, const void* gsrc) {
#ifdef HAVE_ASYNC_LDS
    gv4i* g = (gv4i*)(unsigned long long)gsrc;
    lv4i* l = (lv4i*)(unsigned)(unsigned long long)lds_dst;
    __builtin_amdgcn_global_load_async_to_lds_b128(g, l, 0, 0);
#else
    *(u32x4*)lds_dst = *(const u32x4*)gsrc;
#endif
}

__device__ __forceinline__ void wait_async0() {
#ifdef HAVE_ASYNC_LDS
# if __has_builtin(__builtin_amdgcn_s_wait_asynccnt)
    __builtin_amdgcn_s_wait_asynccnt(0);
# else
    asm volatile("s_wait_asynccnt 0" ::: "memory");
# endif
#endif
}

__device__ __forceinline__ u16 f2bf(float f) {
    unsigned x = __float_as_uint(f);
    return (u16)((x + 0x7FFFu + ((x >> 16) & 1u)) >> 16);
}

// ---------------- block-wide sum over 256 threads ----------------
__device__ __forceinline__ float block_sum256(float v) {
    __shared__ float red[8];
    __shared__ float total;
    for (int off = 16; off > 0; off >>= 1) v += __shfl_xor(v, off, 32);
    __syncthreads();
    if ((threadIdx.x & 31) == 0) red[threadIdx.x >> 5] = v;
    __syncthreads();
    if (threadIdx.x == 0) {
        float s = 0.f;
        for (int i = 0; i < 8; ++i) s += red[i];
        total = s;
    }
    __syncthreads();
    return total;
}

// ---------------- weight transpose + pad to bf16 [Npad x Kpad] ----------------
__global__ void transpose_pad_bf16(const float* __restrict__ W, int K, int N,
                                   u16* __restrict__ Wt, int Kpad, int Npad) {
    int idx = blockIdx.x * 256 + threadIdx.x;
    if (idx >= Npad * Kpad) return;
    int n = idx / Kpad, k = idx - n * Kpad;
    float v = (n < N && k < K) ? W[(size_t)k * N + n] : 0.f;
    Wt[idx] = f2bf(v);
}

// ---------------- rmsnorm -> bf16 (K-padded row) ----------------
__global__ __launch_bounds__(256)
void rmsnorm_bf16_kernel(const float* __restrict__ X, const float* __restrict__ w,
                         u16* __restrict__ Hb) {
    int row = blockIdx.x;
    const float* x = X + (size_t)row * E_;
    float s = 0.f;
    for (int i = threadIdx.x; i < E_; i += 256) { float v = x[i]; s += v * v; }
    float tot = block_sum256(s);
    float rs = rsqrtf(tot / (float)E_ + 1e-6f);
    for (int i = threadIdx.x; i < EP_; i += 256)
        Hb[(size_t)row * EP_ + i] = (i < E_) ? f2bf(x[i] * rs * w[i]) : (u16)0;
}

// ---------------- bf16 WMMA GEMM: C[M x Nreal] = A[M x K] * Bt[N x K]^T ----------------
// block = 256 threads (8 waves); block tile 256 x 64; wave tile 32 x 64.
// B tile (64x32) double-buffered in LDS via async global->LDS copies.
__global__ __launch_bounds__(256)
void gemm_bf16_wmma(const u16* __restrict__ A, int lda,
                    const u16* __restrict__ Bt, int ldb,
                    float* __restrict__ C, int ldc, int Nreal,
                    const float* __restrict__ bias,
                    const float* __restrict__ residual, int ldr,
                    int K) {
    __shared__ __align__(16) u16 lB[2][64][32];
    const int tid  = threadIdx.x;
    const int wave = tid >> 5;
    const int lane = tid & 31;
    const int tileM = blockIdx.x * 256 + wave * 32;
    const int tileN = blockIdx.y * 64;

    f32x8 acc[2][4];
    #pragma unroll
    for (int m = 0; m < 2; ++m)
        #pragma unroll
        for (int j = 0; j < 4; ++j) acc[m][j] = (f32x8){0,0,0,0,0,0,0,0};

    // A fragment addressing (ISA 16-bit A 16x32 layout)
    const int rA  = lane & 15;
    const int khA = (lane >> 4) << 3;       // 0 or 8
    const u16* arow0 = A + (size_t)(tileM + rA) * lda + khA;
    const u16* arow1 = arow0 + (size_t)16 * lda;

    // B fragment addressing (ISA 16-bit layout, same K order as A)
    const int colB = lane & 15;
    const int kbB  = (lane >> 4) << 3;      // 0 or 8

    // staging mapping: each thread copies one 16B chunk of the 64x32 B tile
    const int sn = tid >> 2;
    const int sk = (tid & 3) << 3;
    const u16* bsrc = Bt + (size_t)(tileN + sn) * ldb + sk;

    // prologue: stage first tile into buffer 0
    async_copy16(&lB[0][sn][sk], bsrc);

    int cur = 0;
    for (int k0 = 0; k0 < K; k0 += 32, cur ^= 1) {
        wait_async0();            // this wave's LDS writes landed
        __syncthreads();          // tile visible to all waves; prev reads done
        if (k0 + 32 < K)
            async_copy16(&lB[cur ^ 1][sn][sk], bsrc + k0 + 32);

        Pack32 ap;
        ap.lo = *(const u32x4*)(arow0 + k0);
        ap.hi = *(const u32x4*)(arow0 + k0 + 16);
        bf16x16 av0 = __builtin_bit_cast(bf16x16, ap);
        ap.lo = *(const u32x4*)(arow1 + k0);
        ap.hi = *(const u32x4*)(arow1 + k0 + 16);
        bf16x16 av1 = __builtin_bit_cast(bf16x16, ap);

        #pragma unroll
        for (int j = 0; j < 4; ++j) {
            Pack32 bp;
            bp.lo = *(const u32x4*)&lB[cur][j * 16 + colB][kbB];
            bp.hi = *(const u32x4*)&lB[cur][j * 16 + colB][kbB + 16];
            bf16x16 bv = __builtin_bit_cast(bf16x16, bp);
            acc[0][j] = __builtin_amdgcn_wmma_f32_16x16x32_bf16(
                false, av0, false, bv, (short)0, acc[0][j], false, false);
            acc[1][j] = __builtin_amdgcn_wmma_f32_16x16x32_bf16(
                false, av1, false, bv, (short)0, acc[1][j], false, false);
        }
    }

    // C/D layout: VGPR r -> M = r + 8*(lane>=16), N = lane&15
    const int col0  = lane & 15;
    const int rbase = (lane >> 4) << 3;
    #pragma unroll
    for (int m = 0; m < 2; ++m) {
        #pragma unroll
        for (int j = 0; j < 4; ++j) {
            int col = tileN + j * 16 + col0;
            if (col < Nreal) {
                float badd = bias ? bias[col] : 0.f;
                #pragma unroll
                for (int r = 0; r < 8; ++r) {
                    int row = tileM + m * 16 + rbase + r;
                    float v = acc[m][j][r] + badd;
                    if (residual) v += residual[(size_t)row * ldr + col];
                    C[(size_t)row * ldc + col] = v;
                }
            }
        }
    }
}

// ---------------- compress MLP: per (b,h,lc) block of 32 threads ----------------
__global__ __launch_bounds__(32)
void compress_kernel(const float* __restrict__ Kbuf, const float* __restrict__ Vbuf,
                     const float* __restrict__ w1, const float* __restrict__ b1,
                     const float* __restrict__ w2, const float* __restrict__ b2,
                     float* __restrict__ kc, float* __restrict__ vc) {
    int gid = blockIdx.x;
    int lc = gid & (LC_ - 1);
    int bh = gid >> 7;
    int h = bh % H_, b = bh / H_;
    __shared__ float inb[140];
    __shared__ float hid[10];
    const int t = threadIdx.x;
    for (int src = 0; src < 2; ++src) {
        const float* S = src ? Vbuf : Kbuf;
        float* O = src ? vc : kc;
        for (int i = t; i < 140; i += 32) {
            int cb = i / DH_, d = i - cb * DH_;
            inb[i] = S[((size_t)(b * L_) + lc * CB_ + cb) * E_ + h * DH_ + d];
        }
        __syncthreads();
        if (t < 10) {
            float s = b1[t];
            for (int i = 0; i < 140; ++i) s += inb[i] * w1[i * 10 + t];
            hid[t] = fmaxf(s, 0.f);
        }
        __syncthreads();
        if (t < DH_) {
            float s = b2[t];
            for (int j = 0; j < 10; ++j) s += hid[j] * w2[j * DH_ + t];
            O[((size_t)bh * LC_ + lc) * DH_ + t] = s;
        }
        __syncthreads();
    }
}

// ---------------- compressed attention + block scores: per (b,h) ----------------
__global__ __launch_bounds__(256)
void comp_attn_kernel(const float* __restrict__ Q, const float* __restrict__ kc,
                      const float* __restrict__ vc, float* __restrict__ comp_out,
                      float* __restrict__ bscore) {
    __shared__ float kcs[LC_ * DH_];
    __shared__ float vcs[LC_ * DH_];
    __shared__ float bs[LC_];
    __shared__ float aw[8][LC_];
    int bh = blockIdx.x;
    int h = bh % H_, b = bh / H_;
    int t = threadIdx.x, wave = t >> 5, lane = t & 31;
    for (int i = t; i < LC_ * DH_; i += 256) {
        kcs[i] = kc[(size_t)bh * LC_ * DH_ + i];
        vcs[i] = vc[(size_t)bh * LC_ * DH_ + i];
    }
    if (t < LC_) bs[t] = 0.f;
    __syncthreads();
    for (int l = wave; l < L_; l += 8) {
        const float* q = Q + ((size_t)(b * L_) + l) * E_ + h * DH_;
        float s[4];
        float m = -1e30f;
        #pragma unroll
        for (int j = 0; j < 4; ++j) {
            int key = lane * 4 + j;
            float a = 0.f;
            for (int d = 0; d < DH_; ++d) a += q[d] * kcs[key * DH_ + d];
            s[j] = a * SCALE_;
            m = fmaxf(m, s[j]);
        }
        for (int off = 16; off > 0; off >>= 1) m = fmaxf(m, __shfl_xor(m, off, 32));
        float sum = 0.f;
        #pragma unroll
        for (int j = 0; j < 4; ++j) { s[j] = __expf(s[j] - m); sum += s[j]; }
        for (int off = 16; off > 0; off >>= 1) sum += __shfl_xor(sum, off, 32);
        float inv = 1.f / sum;
        #pragma unroll
        for (int j = 0; j < 4; ++j) {
            float a = s[j] * inv;
            aw[wave][lane * 4 + j] = a;
            atomicAdd(&bs[lane * 4 + j], a);     // ds_add_f32
        }
        asm volatile("s_wait_dscnt 0" ::: "memory");
        if (lane < DH_) {
            float o = 0.f;
            for (int key = 0; key < LC_; ++key) o += aw[wave][key] * vcs[key * DH_ + lane];
            comp_out[((size_t)(b * L_) + l) * E_ + h * DH_ + lane] = o;
        }
    }
    __syncthreads();
    if (t < LC_) bscore[(size_t)bh * LC_ + t] = bs[t];
}

// ---------------- top-k selection (value desc, index asc) ----------------
__global__ void topk_kernel(const float* __restrict__ bscore, int* __restrict__ idx) {
    int bh = blockIdx.x * 64 + threadIdx.x;
    if (bh >= B_ * H_) return;
    const float* s = bscore + (size_t)bh * LC_;
    float pv = 3.4e38f; int pi = -1;
    for (int t = 0; t < TOPK_; ++t) {
        float bestv = -3.4e38f; int besti = 0;
        for (int k = 0; k < LC_; ++k) {
            float v = s[k];
            bool after_prev = (v < pv) || (v == pv && k > pi);
            if (after_prev && v > bestv) { bestv = v; besti = k; }
        }
        idx[bh * TOPK_ + t] = besti;
        pv = bestv; pi = besti;
    }
}

// ---------------- selected + window attention, gate, combine: per (b,h) ----------------
__global__ __launch_bounds__(256)
void selwin_kernel(const float* __restrict__ Q, const float* __restrict__ Kbuf,
                   const float* __restrict__ Vbuf, const int* __restrict__ idx,
                   const float* __restrict__ comp_out,
                   const float* __restrict__ gate_w, const float* __restrict__ gate_b,
                   float* __restrict__ attn_out) {
    __shared__ float ks[32 * DH_], vs[32 * DH_];
    __shared__ float kw[WS_ * DH_], vw[WS_ * DH_];
    __shared__ float asel[8][32], awin[8][WS_];
    __shared__ float gw[DH_ * 3], gb[3];
    int bh = blockIdx.x;
    int h = bh % H_, b = bh / H_;
    int t = threadIdx.x, wave = t >> 5, lane = t & 31;
    for (int i = t; i < 32 * DH_; i += 256) {
        int key = i / DH_, d = i - key * DH_;
        int blk = idx[bh * TOPK_ + (key >> 1)];
        int l = blk * SB_ + (key & 1);
        size_t off = ((size_t)(b * L_) + l) * E_ + h * DH_ + d;
        ks[i] = Kbuf[off]; vs[i] = Vbuf[off];
    }
    for (int i = t; i < WS_ * DH_; i += 256) {
        int key = i / DH_, d = i - key * DH_;
        size_t off = ((size_t)(b * L_) + (L_ - WS_ + key)) * E_ + h * DH_ + d;
        kw[i] = Kbuf[off]; vw[i] = Vbuf[off];
    }
    if (t < DH_ * 3) gw[t] = gate_w[t];
    if (t < 3) gb[t] = gate_b[t];
    __syncthreads();
    for (int l = wave; l < L_; l += 8) {
        const float* q = Q + ((size_t)(b * L_) + l) * E_ + h * DH_;
        // selected branch: one key per lane
        float s = 0.f;
        for (int d = 0; d < DH_; ++d) s += q[d] * ks[lane * DH_ + d];
        s *= SCALE_;
        float m = s;
        for (int off = 16; off > 0; off >>= 1) m = fmaxf(m, __shfl_xor(m, off, 32));
        float e = __expf(s - m);
        float sum = e;
        for (int off = 16; off > 0; off >>= 1) sum += __shfl_xor(sum, off, 32);
        asel[wave][lane] = e / sum;
        // window branch: 5 keys on lanes 0..4
        float swv = -1e30f;
        if (lane < WS_) {
            float a = 0.f;
            for (int d = 0; d < DH_; ++d) a += q[d] * kw[lane * DH_ + d];
            swv = a * SCALE_;
        }
        float mw = swv;
        for (int off = 16; off > 0; off >>= 1) mw = fmaxf(mw, __shfl_xor(mw, off, 32));
        float ew = (lane < WS_) ? __expf(swv - mw) : 0.f;
        float sumw = ew;
        for (int off = 16; off > 0; off >>= 1) sumw += __shfl_xor(sumw, off, 32);
        if (lane < WS_) awin[wave][lane] = ew / sumw;
        asm volatile("s_wait_dscnt 0" ::: "memory");
        if (lane < DH_) {
            int d = lane;
            float osel = 0.f, owin = 0.f;
            for (int k2 = 0; k2 < 32; ++k2)  osel += asel[wave][k2] * vs[k2 * DH_ + d];
            for (int k2 = 0; k2 < WS_; ++k2) owin += awin[wave][k2] * vw[k2 * DH_ + d];
            float g0 = gb[0], g1 = gb[1], g2 = gb[2];
            for (int dd = 0; dd < DH_; ++dd) {
                float qv = q[dd];
                g0 += qv * gw[dd * 3 + 0];
                g1 += qv * gw[dd * 3 + 1];
                g2 += qv * gw[dd * 3 + 2];
            }
            float gm = fmaxf(g0, fmaxf(g1, g2));
            float e0 = __expf(g0 - gm), e1 = __expf(g1 - gm), e2 = __expf(g2 - gm);
            float gi = 1.f / (e0 + e1 + e2);
            size_t off = ((size_t)(b * L_) + l) * E_ + h * DH_ + d;
            attn_out[off] = gi * (e0 * comp_out[off] + e1 * osel + e2 * owin);
        }
    }
}

// ---------------- elementwise helpers ----------------
__global__ void add_kernel(const float* __restrict__ a, const float* __restrict__ b,
                           float* __restrict__ o, int n) {
    int i = blockIdx.x * 256 + threadIdx.x;
    if (i < n) o[i] = a[i] + b[i];
}

__global__ void silu_mul_kernel(const float* __restrict__ gt, const float* __restrict__ up,
                                u16* __restrict__ act, int n) {
    int i = blockIdx.x * 256 + threadIdx.x;
    if (i < n) {
        float g = gt[i];
        float s = g / (1.f + __expf(-g));
        act[i] = f2bf(s * up[i]);
    }
}

// ---------------- final double layernorm over (x + y) ----------------
__global__ __launch_bounds__(256)
void final_norm_kernel(const float* __restrict__ X, const float* __restrict__ Y,
                       const float* __restrict__ w1, const float* __restrict__ b1,
                       const float* __restrict__ w2, const float* __restrict__ b2,
                       float* __restrict__ out) {
    __shared__ float buf[E_];
    int row = blockIdx.x, t = threadIdx.x;
    const float* x = X + (size_t)row * E_;
    const float* y = Y + (size_t)row * E_;
    float s = 0.f;
    for (int i = t; i < E_; i += 256) { float v = x[i] + y[i]; buf[i] = v; s += v; }
    float mean = block_sum256(s) / (float)E_;
    float s2 = 0.f;
    for (int i = t; i < E_; i += 256) { float d = buf[i] - mean; s2 += d * d; }
    float inv = rsqrtf(block_sum256(s2) / (float)E_ + 1e-6f);
    for (int i = t; i < E_; i += 256) buf[i] = (buf[i] - mean) * inv * w1[i] + b1[i];
    __syncthreads();
    float s3 = 0.f;
    for (int i = t; i < E_; i += 256) s3 += buf[i];
    float mean2 = block_sum256(s3) / (float)E_;
    float s4 = 0.f;
    for (int i = t; i < E_; i += 256) { float d = buf[i] - mean2; s4 += d * d; }
    float inv2 = rsqrtf(block_sum256(s4) / (float)E_ + 1e-6f);
    for (int i = t; i < E_; i += 256)
        out[(size_t)row * E_ + i] = (buf[i] - mean2) * inv2 * w2[i] + b2[i];
}

// ===================================================================
extern "C" void kernel_launch(void* const* d_in, const int* in_sizes, int n_in,
                              void* d_out, int out_size, void* d_ws, size_t ws_size,
                              hipStream_t stream) {
    (void)in_sizes; (void)n_in; (void)out_size; (void)ws_size;
    const float* x        = (const float*)d_in[0];
    const float* anw      = (const float*)d_in[1];
    const float* q_w      = (const float*)d_in[2];
    const float* q_b      = (const float*)d_in[3];
    const float* k_w      = (const float*)d_in[4];
    const float* k_b      = (const float*)d_in[5];
    const float* v_w      = (const float*)d_in[6];
    const float* v_b      = (const float*)d_in[7];
    const float* comp_w1  = (const float*)d_in[8];
    const float* comp_b1  = (const float*)d_in[9];
    const float* comp_w2  = (const float*)d_in[10];
    const float* comp_b2  = (const float*)d_in[11];
    const float* gate_w   = (const float*)d_in[12];
    const float* gate_b   = (const float*)d_in[13];
    const float* mnw      = (const float*)d_in[14];
    const float* g_gate_w = (const float*)d_in[15];
    const float* g_up_w   = (const float*)d_in[16];
    const float* g_down_w = (const float*)d_in[17];
    const float* ln1_w    = (const float*)d_in[18];
    const float* ln1_b    = (const float*)d_in[19];
    const float* ln2_w    = (const float*)d_in[20];
    const float* ln2_b    = (const float*)d_in[21];
    float* out = (float*)d_out;

    char* ws = (char*)d_ws;
    size_t off = 0;
    auto alloc = [&](size_t bytes) -> void* {
        void* p = ws + off;
        off = (off + bytes + 255) & ~(size_t)255;
        return p;
    };
    u16*   hb   = (u16*)alloc((size_t)M_ * EP_ * 2);
    u16*   h2b  = (u16*)alloc((size_t)M_ * EP_ * 2);
    u16*   actb = (u16*)alloc((size_t)M_ * IM_ * 2);
    u16*   WqT  = (u16*)alloc((size_t)EP_ * EP_ * 2);
    u16*   WkT  = (u16*)alloc((size_t)EP_ * EP_ * 2);
    u16*   WvT  = (u16*)alloc((size_t)EP_ * EP_ * 2);
    u16*   WgT  = (u16*)alloc((size_t)IM_ * EP_ * 2);
    u16*   WuT  = (u16*)alloc((size_t)IM_ * EP_ * 2);
    u16*   WdT  = (u16*)alloc((size_t)EP_ * IM_ * 2);
    float* qb   = (float*)alloc((size_t)M_ * E_ * 4);
    float* kb   = (float*)alloc((size_t)M_ * E_ * 4);
    float* vb   = (float*)alloc((size_t)M_ * E_ * 4);
    float* kc   = (float*)alloc((size_t)B_ * H_ * LC_ * DH_ * 4);
    float* vc   = (float*)alloc((size_t)B_ * H_ * LC_ * DH_ * 4);
    float* bsc  = (float*)alloc((size_t)B_ * H_ * LC_ * 4);
    int*   tk   = (int*)alloc((size_t)B_ * H_ * TOPK_ * 4);
    float* cout = (float*)alloc((size_t)M_ * E_ * 4);
    float* aout = (float*)alloc((size_t)M_ * E_ * 4);
    float* x1   = (float*)alloc((size_t)M_ * E_ * 4);
    float* gt   = (float*)alloc((size_t)M_ * IM_ * 4);
    float* up   = (float*)alloc((size_t)M_ * IM_ * 4);
    float* yb   = (float*)alloc((size_t)M_ * E_ * 4);

    // 1) weight transpose+pad to bf16
    {
        int n1 = EP_ * EP_, g1 = (n1 + 255) / 256;
        transpose_pad_bf16<<<g1, 256, 0, stream>>>(q_w, E_, E_, WqT, EP_, EP_);
        transpose_pad_bf16<<<g1, 256, 0, stream>>>(k_w, E_, E_, WkT, EP_, EP_);
        transpose_pad_bf16<<<g1, 256, 0, stream>>>(v_w, E_, E_, WvT, EP_, EP_);
        int n2 = IM_ * EP_, g2 = (n2 + 255) / 256;
        transpose_pad_bf16<<<g2, 256, 0, stream>>>(g_gate_w, E_, IM_, WgT, EP_, IM_);
        transpose_pad_bf16<<<g2, 256, 0, stream>>>(g_up_w,   E_, IM_, WuT, EP_, IM_);
        int n3 = EP_ * IM_, g3 = (n3 + 255) / 256;
        transpose_pad_bf16<<<g3, 256, 0, stream>>>(g_down_w, IM_, E_, WdT, IM_, EP_);
    }

    // 2) h = rmsnorm(x) -> bf16
    rmsnorm_bf16_kernel<<<M_, 256, 0, stream>>>(x, anw, hb);

    // 3) q,k,v projections (WMMA)
    dim3 gqkv(M_ / 256, EP_ / 64);
    gemm_bf16_wmma<<<gqkv, 256, 0, stream>>>(hb, EP_, WqT, EP_, qb, E_, E_, q_b, nullptr, 0, EP_);
    gemm_bf16_wmma<<<gqkv, 256, 0, stream>>>(hb, EP_, WkT, EP_, kb, E_, E_, k_b, nullptr, 0, EP_);
    gemm_bf16_wmma<<<gqkv, 256, 0, stream>>>(hb, EP_, WvT, EP_, vb, E_, E_, v_b, nullptr, 0, EP_);

    // 4) compress k,v
    compress_kernel<<<B_ * H_ * LC_, 32, 0, stream>>>(kb, vb, comp_w1, comp_b1,
                                                      comp_w2, comp_b2, kc, vc);

    // 5) compressed attention + block scores
    comp_attn_kernel<<<B_ * H_, 256, 0, stream>>>(qb, kc, vc, cout, bsc);

    // 6) top-k blocks
    topk_kernel<<<(B_ * H_ + 63) / 64, 64, 0, stream>>>(bsc, tk);

    // 7) selected + window attention, gate, combine
    selwin_kernel<<<B_ * H_, 256, 0, stream>>>(qb, kb, vb, tk, cout, gate_w, gate_b, aout);

    // 8) x1 = x + attn_out
    add_kernel<<<(M_ * E_ + 255) / 256, 256, 0, stream>>>(x, aout, x1, M_ * E_);

    // 9) h2 = rmsnorm(x1) -> bf16
    rmsnorm_bf16_kernel<<<M_, 256, 0, stream>>>(x1, mnw, h2b);

    // 10) gate & up projections (WMMA)
    dim3 gmlp(M_ / 256, IM_ / 64);
    gemm_bf16_wmma<<<gmlp, 256, 0, stream>>>(h2b, EP_, WgT, EP_, gt, IM_, IM_, nullptr, nullptr, 0, EP_);
    gemm_bf16_wmma<<<gmlp, 256, 0, stream>>>(h2b, EP_, WuT, EP_, up, IM_, IM_, nullptr, nullptr, 0, EP_);

    // 11) act = silu(gt) * up -> bf16
    silu_mul_kernel<<<(M_ * IM_ + 255) / 256, 256, 0, stream>>>(gt, up, actb, M_ * IM_);

    // 12) y = x1 + act @ down_w (WMMA, residual epilogue)
    gemm_bf16_wmma<<<gqkv, 256, 0, stream>>>(actb, IM_, WdT, IM_, yb, E_, E_, nullptr, x1, E_, IM_);

    // 13) out = layernorm(layernorm(x + y, ln1), ln2)
    final_norm_kernel<<<M_, 256, 0, stream>>>(x, yb, ln1_w, ln1_b, ln2_w, ln2_b, out);
}
